// KimiLinearKDAPrefill_42803644072881
// MI455X (gfx1250) — matrix operand order repaired and checked by reading, not compiled, not run on hardware
//
#include <hip/hip_runtime.h>
#include <hip/hip_bf16.h>

#define T_TOK 8192
#define H_N   16
#define D_DIM 128
#define HD    (H_N * D_DIM)      /* 2048 */
#define C3    (3 * HD)           /* 6144 */
#define KW    4
#define CHUNK 64
#define NCHUNK (T_TOK / CHUNK)   /* 128 */
#define DVB   32                 /* v-columns per workgroup */
#define EPS_F 1e-6f

typedef __attribute__((ext_vector_type(16))) __bf16 bf16x16;
typedef __attribute__((ext_vector_type(8)))  float  f32x8;

// ---------------------------------------------------------------------------
// WMMA fragment helpers (wave32, 16x16x32 bf16).
// A operand: stored row-major [M][K]. Per ISA 7.12.2: lanes 0-15 row m, k-base 0;
// lanes 16-31 row m-16... here rows 0-15 with k {0..7,16..23} / {8..15,24..31}.
// B operand: stored transposed [N][K]; lane n holds k {0..7,16..23} (lanes 0-15) or
// {8..15,24..31} (lanes 16-31) of column n.
// ---------------------------------------------------------------------------
__device__ __forceinline__ bf16x16 frag_A(const __bf16* base, int ld, int m0, int k0) {
  const int lane = threadIdx.x & 31;
  const int m  = m0 + (lane & 15);
  const int kb = k0 + ((lane >> 4) << 3);          // 0 or 8
  const uint4* p = reinterpret_cast<const uint4*>(base + m * ld + kb);
  union { uint4 u[2]; bf16x16 v; } f;
  f.u[0] = p[0];                                   // k = kb .. kb+7
  f.u[1] = p[2];                                   // k = kb+16 .. kb+23
  return f.v;
}

__device__ __forceinline__ bf16x16 frag_B(const __bf16* baseT, int ld, int n0, int k0) {
  const int lane = threadIdx.x & 31;
  const int n  = n0 + (lane & 15);
  const int kb = k0 + ((lane >> 4) << 3);          // 0 or 8
  const uint4* p = reinterpret_cast<const uint4*>(baseT + n * ld + kb);
  union { uint4 u[2]; bf16x16 v; } f;
  f.u[0] = p[0];
  f.u[1] = p[2];
  return f.v;
}

__device__ __forceinline__ f32x8 wmma_bf16(bf16x16 a, bf16x16 b, f32x8 c) {
  return __builtin_amdgcn_wmma_f32_16x16x32_bf16(false, a, false, b, (short)0, c,
                                                 false, false);
}

// ---------------------------------------------------------------------------
// Kernel 1: causal conv1d(K=4) + SiLU, per-head l2norm on q/k, KDA gate, sigmoid.
// One workgroup per token. Bandwidth bound.
// ---------------------------------------------------------------------------
__global__ __launch_bounds__(256)
void kda_preprocess(const float* __restrict__ mixed, const float* __restrict__ fg,
                    const float* __restrict__ beta,  const float* __restrict__ cw,
                    const float* __restrict__ dtb,   const float* __restrict__ alog,
                    __bf16* __restrict__ qb, __bf16* __restrict__ kb,
                    __bf16* __restrict__ vb, float* __restrict__ gbuf,
                    float* __restrict__ bs) {
  const int t   = blockIdx.x;
  const int tid = threadIdx.x;
  __shared__ float xb[C3];
  __shared__ float red[32][8];
  __shared__ float nrm[32];

  for (int c = tid; c < C3; c += 256) {
    float acc = 0.f;
#pragma unroll
    for (int j = 0; j < KW; ++j) {
      const int tt = t - (KW - 1) + j;
      if (tt >= 0) acc += mixed[(size_t)tt * C3 + c] * cw[c * KW + j];
    }
    xb[c] = acc / (1.f + __expf(-acc));            // SiLU
  }
  __syncthreads();
  {                                                // q/k row sums of squares
    const int row = tid >> 3, part = tid & 7;      // 32 rows x 8 partials
    const float* p = &xb[row * D_DIM + part * 16];
    float s = 0.f;
#pragma unroll
    for (int i = 0; i < 16; ++i) s += p[i] * p[i];
    red[row][part] = s;
  }
  __syncthreads();
  if (tid < 32) {
    float s = 0.f;
#pragma unroll
    for (int i = 0; i < 8; ++i) s += red[tid][i];
    nrm[tid] = rsqrtf(s + EPS_F);
  }
  __syncthreads();
  const float qscale = 0.08838834764831845f;       // D^-0.5
  for (int c = tid; c < HD; c += 256) {
    const int hh = c >> 7;
    qb[(size_t)t * HD + c] = (__bf16)(xb[c] * nrm[hh] * qscale);
    kb[(size_t)t * HD + c] = (__bf16)(xb[HD + c] * nrm[16 + hh]);
    vb[(size_t)t * HD + c] = (__bf16)xb[2 * HD + c];
    const float raw = fg[(size_t)t * HD + c] + dtb[c];
    const float sp  = (raw > 20.f) ? raw : log1pf(__expf(raw));
    gbuf[(size_t)t * HD + c] = -__expf(alog[hh]) * sp;   // log-decay (<= 0)
  }
  if (tid < H_N)
    bs[(size_t)t * H_N + tid] = 1.f / (1.f + __expf(-beta[(size_t)t * H_N + tid]));
}

// ---------------------------------------------------------------------------
// Kernel 2: chunked gated delta rule. grid = (D/DVB, H); 256 threads (8 waves).
// ---------------------------------------------------------------------------
__global__ __launch_bounds__(256)
void kda_chunk_scan(const __bf16* __restrict__ qb, const __bf16* __restrict__ kb,
                    const __bf16* __restrict__ vb, const float* __restrict__ gbuf,
                    const float* __restrict__ bs,  float* __restrict__ out) {
  const int h    = blockIdx.y;
  const int dv0  = blockIdx.x * DVB;
  const int tid  = threadIdx.x;
  const int w    = tid >> 5;
  const int lane = tid & 31;
  const int rbase = (lane >> 4) << 3;              // C-fragment row base (0/8)
  const int ncol  = lane & 15;                     // C-fragment column

  __shared__ __align__(16) __bf16 sW[CHUNK][D_DIM];     // k * exp(Gc)
  __shared__ __align__(16) __bf16 sKinv[CHUNK][D_DIM];  // k * exp(-Gc) (clamped)
  __shared__ __align__(16) __bf16 sQt[CHUNK][D_DIM];    // q * exp(Gc)
  __shared__ __align__(16) __bf16 sKdT[D_DIM][CHUNK];   // (k * exp(Gend-Gc))^T
  __shared__ __align__(16) __bf16 sS0T[DVB][D_DIM];     // bf16 S0^T for matmuls
  __shared__ __align__(16) float  sS[D_DIM][DVB];       // f32 running state
  __shared__ __align__(16) float  sDelta[CHUNK][DVB];   // RHS -> solved delta
  __shared__ __align__(16) __bf16 sDT[DVB][CHUNK];      // delta^T bf16 (zero-padded)
  __shared__ float sB[CHUNK];
  __shared__ float sGend[D_DIM];                        // log total decay
  union Scr {
    float gc[CHUNK][D_DIM];                             // cumsum(g) (32 KB) ...
    struct {                                            // ... overlaid with:
      float  m[CHUNK][CHUNK];                           // BM (strict lower) f32
      __bf16 mb[CHUNK][CHUNK];                          // BM bf16
      __bf16 aqo[CHUNK][CHUNK];                         // masked q-k attn bf16
    } mm;
  };
  __shared__ __align__(16) Scr u;

  for (int i = tid; i < D_DIM * DVB; i += 256) ((float*)sS)[i] = 0.f;

  for (int ci = 0; ci < NCHUNK; ++ci) {
    const int t0 = ci * CHUNK;
    __syncthreads();                               // prev chunk fully done

    // --- stage 0: S0^T -> bf16, zero deltaT, load beta, prefetch next gates ---
    for (int i = tid; i < D_DIM * DVB; i += 256) {
      const int d = i >> 5, dv = i & 31;
      sS0T[dv][d] = (__bf16)sS[d][dv];
    }
    for (int i = tid; i < DVB * CHUNK; i += 256) ((__bf16*)sDT)[i] = (__bf16)0.f;
    if (tid < CHUNK) sB[tid] = bs[(size_t)(t0 + tid) * H_N + h];
    if (ci + 1 < NCHUNK && tid >= 128) {           // gfx1250 global_prefetch_b8
      const int tt = tid - 128;
      __builtin_prefetch(gbuf + (size_t)(t0 + CHUNK + (tt >> 1)) * HD + h * D_DIM
                              + (tt & 1) * 64, 0, 1);
    }
    // --- stage 1: inclusive cumsum of log-decay over time (one lane per dim) --
    if (tid < D_DIM) {
      float acc = 0.f;
      const float* pg = gbuf + (size_t)t0 * HD + h * D_DIM + tid;
      for (int t = 0; t < CHUNK; ++t) {
        acc += pg[(size_t)t * HD];
        u.gc[t][tid] = acc;
      }
      sGend[tid] = acc;                            // log(total chunk decay)
    }
    __syncthreads();
    // --- stage 2: build bf16 operand tiles ----------------------------------
    for (int i = tid; i < CHUNK * D_DIM; i += 256) {
      const int t = i >> 7, d = i & 127;
      const float gc  = u.gc[t][d];
      const float eg  = __expf(gc);
      const float egi = __expf(fminf(-gc, 30.f));
      const size_t gi = (size_t)(t0 + t) * HD + h * D_DIM + d;
      const float kf = (float)kb[gi];
      const float qf = (float)qb[gi];
      sW[t][d]    = (__bf16)(kf * eg);
      sKinv[t][d] = (__bf16)(kf * egi);
      sQt[t][d]   = (__bf16)(qf * eg);
      sKdT[d][t]  = (__bf16)(kf * __expf(sGend[d] - gc));
    }
    __syncthreads();
    // --- stage 3: M = W Kinv^T (+beta, strict-lower), Aqo = Qt Kinv^T (incl), -
    //              U0 = W S0 -> RHS P = B(V - U0) -------------------------------
    for (int tt = 2 * w; tt < 2 * w + 2; ++tt) {   // 16 tiles of 64x64, 2/wave
      const int mi = tt >> 2, ni = tt & 3;
      f32x8 c = {};
      f32x8 c2 = {};
#pragma unroll
      for (int kk = 0; kk < 4; ++kk) {
        const bf16x16 bfr = frag_B(&sKinv[0][0], D_DIM, ni * 16, kk * 32);
        c  = wmma_bf16(frag_A(&sW[0][0],  D_DIM, mi * 16, kk * 32), bfr, c);
        c2 = wmma_bf16(frag_A(&sQt[0][0], D_DIM, mi * 16, kk * 32), bfr, c2);
      }
#pragma unroll
      for (int r = 0; r < 8; ++r) {
        const int m = mi * 16 + rbase + r;
        const int n = ni * 16 + ncol;
        const float bm = (m > n) ? sB[m] * c[r] : 0.f;
        u.mm.m[m][n]  = bm;
        u.mm.mb[m][n] = (__bf16)bm;
        u.mm.aqo[m][n] = (__bf16)((m >= n) ? c2[r] : 0.f);
      }
    }
    {                                              // 8 tiles of 64x32, 1/wave
      const int mi = w >> 1, ni = w & 1;
      f32x8 c = {};
#pragma unroll
      for (int kk = 0; kk < 4; ++kk)
        c = wmma_bf16(frag_A(&sW[0][0],   D_DIM, mi * 16, kk * 32),
                      frag_B(&sS0T[0][0], D_DIM, ni * 16, kk * 32), c);
#pragma unroll
      for (int r = 0; r < 8; ++r) {
        const int m = mi * 16 + rbase + r;
        const int n = ni * 16 + ncol;
        const float vf = (float)vb[(size_t)(t0 + m) * HD + h * D_DIM + dv0 + n];
        sDelta[m][n] = sB[m] * (vf - c[r]);
      }
    }
    __syncthreads();
    // --- stage 4: blocked forward substitution (I + BM) delta = P -------------
    for (int blk = 0; blk < 4; ++blk) {
      if (blk > 0 && w < 2) {                      // WMMA correction; sDT rows of
        const int ni = w;                          // unsolved blocks are zero
        f32x8 c = {};
        c = wmma_bf16(frag_A(&u.mm.mb[0][0], CHUNK, blk * 16, 0),
                      frag_B(&sDT[0][0],     CHUNK, ni * 16, 0), c);
        c = wmma_bf16(frag_A(&u.mm.mb[0][0], CHUNK, blk * 16, 32),
                      frag_B(&sDT[0][0],     CHUNK, ni * 16, 32), c);
#pragma unroll
        for (int r = 0; r < 8; ++r)
          sDelta[blk * 16 + rbase + r][ni * 16 + ncol] -= c[r];
      }
      __syncthreads();
      if (w == 0) {                                // 16x16 unit-lower solve, f32
        for (int r = blk * 16; r < blk * 16 + 16; ++r) {
          float a = sDelta[r][lane];
          for (int c2i = blk * 16; c2i < r; ++c2i)
            a -= u.mm.m[r][c2i] * sDelta[c2i][lane];
          sDelta[r][lane] = a;
          sDT[lane][r] = (__bf16)a;
        }
      }
      __syncthreads();
    }
    // --- stage 5: O = Qt S0 + Aqo delta --------------------------------------
    {
      const int mi = w >> 1, ni = w & 1;
      f32x8 c = {};
#pragma unroll
      for (int kk = 0; kk < 4; ++kk)
        c = wmma_bf16(frag_A(&sQt[0][0],  D_DIM, mi * 16, kk * 32),
                      frag_B(&sS0T[0][0], D_DIM, ni * 16, kk * 32), c);
#pragma unroll
      for (int kk = 0; kk < 2; ++kk)
        c = wmma_bf16(frag_A(&u.mm.aqo[0][0], CHUNK, mi * 16, kk * 32),
                      frag_B(&sDT[0][0],      CHUNK, ni * 16, kk * 32), c);
#pragma unroll
      for (int r = 0; r < 8; ++r) {
        const int m = mi * 16 + rbase + r;
        const int n = ni * 16 + ncol;
        out[(size_t)(t0 + m) * HD + h * D_DIM + dv0 + n] = c[r];
      }
    }
    // --- stage 6: S = diag(exp(Gend)) S + Kd^T delta -------------------------
    for (int tt = 2 * w; tt < 2 * w + 2; ++tt) {   // 16 tiles of 128x32
      const int mi = tt >> 1, ni = tt & 1;
      f32x8 c = {};
      c = wmma_bf16(frag_A(&sKdT[0][0], CHUNK, mi * 16, 0),
                    frag_B(&sDT[0][0],  CHUNK, ni * 16, 0), c);
      c = wmma_bf16(frag_A(&sKdT[0][0], CHUNK, mi * 16, 32),
                    frag_B(&sDT[0][0],  CHUNK, ni * 16, 32), c);
#pragma unroll
      for (int r = 0; r < 8; ++r) {
        const int m = mi * 16 + rbase + r;
        const int n = ni * 16 + ncol;
        sS[m][n] = __expf(sGend[m]) * sS[m][n] + c[r];
      }
    }
  }
}

// ---------------------------------------------------------------------------
extern "C" void kernel_launch(void* const* d_in, const int* in_sizes, int n_in,
                              void* d_out, int out_size, void* d_ws, size_t ws_size,
                              hipStream_t stream) {
  const float* mixed = (const float*)d_in[0];
  const float* fg    = (const float*)d_in[1];
  const float* beta  = (const float*)d_in[2];
  const float* cw    = (const float*)d_in[3];
  const float* dtb   = (const float*)d_in[4];
  const float* alog  = (const float*)d_in[5];
  float* out = (float*)d_out;

  const size_t nTHD = (size_t)T_TOK * HD;
  char* ws = (char*)d_ws;
  size_t off = 0;
  auto take = [&](size_t bytes) {
    void* p = ws + off;
    off = (off + bytes + 255) & ~(size_t)255;
    return p;
  };
  __bf16* qb = (__bf16*)take(nTHD * sizeof(__bf16));
  __bf16* kb = (__bf16*)take(nTHD * sizeof(__bf16));
  __bf16* vb = (__bf16*)take(nTHD * sizeof(__bf16));
  float*  gb = (float*)take(nTHD * sizeof(float));
  float*  bs = (float*)take((size_t)T_TOK * H_N * sizeof(float));
  if (off > ws_size) return;

  kda_preprocess<<<T_TOK, 256, 0, stream>>>(mixed, fg, beta, cw, dtb, alog,
                                            qb, kb, vb, gb, bs);
  dim3 grid(D_DIM / DVB, H_N);
  kda_chunk_scan<<<grid, 256, 0, stream>>>(qb, kb, vb, gb, bs, out);
}
